// SpatialAttention2d_8169027797511
// MI455X (gfx1250) — matrix operand order repaired and checked by reading, not compiled, not run on hardware
//
#include <hip/hip_runtime.h>
#include <hip/hip_bf16.h>

typedef __attribute__((ext_vector_type(16))) _Float16 v16h;
typedef __attribute__((ext_vector_type(8)))  float    v8f;

#define C_CH  256
#define HW    4096
#define BATCH 8

// ---------------- init per-batch k-range accumulators -------------------
__global__ void init_kernel(unsigned* __restrict__ zmax, unsigned* __restrict__ zmin) {
    int t = threadIdx.x;
    if (t < BATCH) { zmax[t] = 0u; zmin[t] = 0x7F800000u; }  // z >= 0 (ReLU)
}

// ---------------- squeeze: z = relu(x . w_sq), plus min/max(z) ----------
__global__ void squeeze_kernel(const float* __restrict__ x,
                               const float* __restrict__ w_sq,
                               float* __restrict__ z,
                               unsigned* __restrict__ zmax,
                               unsigned* __restrict__ zmin) {
    const int b  = blockIdx.y;
    const int hw = blockIdx.x * blockDim.x + threadIdx.x;
    const float* xp = x + (size_t)b * C_CH * HW + hw;
    float acc = 0.f;
    #pragma unroll 8
    for (int c = 0; c < C_CH; ++c)
        acc = fmaf(xp[(size_t)c * HW], w_sq[c], acc);
    const float zz = fmaxf(acc, 0.f);
    z[b * HW + hw] = zz;

    // z >= 0, so uint bit-pattern order == float order
    __shared__ unsigned smax, smin;
    if (threadIdx.x == 0) { smax = 0u; smin = 0x7F800000u; }
    __syncthreads();
    atomicMax(&smax, __float_as_uint(zz));
    atomicMin(&smin, __float_as_uint(zz));
    __syncthreads();
    if (threadIdx.x == 0) {
        atomicMax(&zmax[b], smax);
        atomicMin(&zmin[b], smin);
    }
}

// ---------------- rank-1 attention via WMMA -----------------------------
// Per wave: one 16-row i-tile. A[16x32 f16] = exp scores, B[32x16 f16] =
// [v_j | 1 | 0...], C/D f32 accumulates numerator (N=0) and denominator (N=1).
__global__ void attn_kernel(const float* __restrict__ z,
                            const float* __restrict__ wq, const float* __restrict__ bq,
                            const float* __restrict__ wk, const float* __restrict__ bk,
                            const float* __restrict__ wv, const float* __restrict__ bv,
                            const float* __restrict__ gamma,
                            const unsigned* __restrict__ zmax,
                            const unsigned* __restrict__ zmin,
                            float* __restrict__ gate) {
    __shared__ float    zs[HW];        // 16 KB
    __shared__ _Float16 vh[HW];        // 8 KB
    __shared__ float    red[8][32];    // per-wave num/den staging

    const int b   = blockIdx.y;
    const int tid = threadIdx.x;
    const float wqv = wq[0], bqv = bq[0];
    const float wkv = wk[0], bkv = bk[0];
    const float wvv = wv[0], bvv = bv[0];
    const float gmv = gamma[0];

    for (int t = tid; t < HW; t += blockDim.x) {
        const float zz = z[b * HW + t];
        zs[t] = zz;
        vh[t] = (_Float16)fmaf(wvv, zz, bvv);
    }
    __syncthreads();

    // row max of q_i*k_j from per-batch z range (k is affine in z)
    const float zmx = __uint_as_float(zmax[b]);
    const float zmn = __uint_as_float(zmin[b]);
    const float ka = fmaf(wkv, zmx, bkv);
    const float kb2 = fmaf(wkv, zmn, bkv);
    const float kmaxv = fmaxf(ka, kb2);
    const float kminv = fminf(ka, kb2);

    const int wave  = tid >> 5;
    const int lane  = tid & 31;
    const int half  = lane >> 4;       // 0: K 0-7/16-23, 1: K 8-15/24-31
    const int row   = lane & 15;       // M within tile
    const int itile = blockIdx.x * 128 + wave * 16;

    const float zi = zs[itile + row];
    const float qv = fmaf(wqv, zi, bqv);
    const float mi = (qv >= 0.f) ? qv * kmaxv : qv * kminv;
    // score = q*(wk*z+bk) - m = alpha*z + beta  -> one FMA per element
    const float alpha = qv * wkv;
    const float beta  = fmaf(qv, bkv, -mi);

    v8f c = {0.f, 0.f, 0.f, 0.f, 0.f, 0.f, 0.f, 0.f};
    const int kb = half * 8;
    for (int jb = 0; jb < HW; jb += 32) {
        v16h a;
        #pragma unroll
        for (int e = 0; e < 8; ++e) {
            a[e]     = (_Float16)__expf(fmaf(alpha, zs[jb + kb + e],      beta));
            a[e + 8] = (_Float16)__expf(fmaf(alpha, zs[jb + 16 + kb + e], beta));
        }
        v16h bf = {};                  // B[k(h,e), n=lane&15], table k order
        if (row == 0) {                // column 0: v_j
            #pragma unroll
            for (int e = 0; e < 8; ++e) {
                bf[e]     = vh[jb + kb + e];
                bf[e + 8] = vh[jb + 16 + kb + e];
            }
        } else if (row == 1) {         // column 1: ones (denominator)
            #pragma unroll
            for (int e = 0; e < 16; ++e) bf[e] = (_Float16)1.0f;
        }
        c = __builtin_amdgcn_wmma_f32_16x16x32_f16(
                false, a, false, bf, (short)0, c, false, false);
    }

    // D layout: VGPR r -> M = r + 8*half, N = lane&15. Keep N=0 (num), N=1 (den).
    const int n = lane & 15;
    if (n < 2) {
        #pragma unroll
        for (int r = 0; r < 8; ++r)
            red[wave][n * 16 + r + half * 8] = c[r];
    }
    __syncthreads();
    if (lane < 16) {
        const float num = red[wave][lane];
        const float den = red[wave][16 + lane];
        const float o = num / den;
        gate[b * HW + itile + lane] = 1.f / (1.f + __expf(-gmv * o));
    }
}

// ---------------- apply: out = x * gate (float4 vectorized) -------------
__global__ void apply_kernel(const float4* __restrict__ x4,
                             const float4* __restrict__ g4,
                             float4* __restrict__ o4) {
    const int i = blockIdx.x * blockDim.x + threadIdx.x;   // over B*C*HW/4
    const float4 xv = x4[i];
    const int b = i >> 18;             // / (C_CH * HW/4)
    const int r = i & 1023;            // float4 index within hw row
    const float4 gv = g4[b * (HW / 4) + r];
    float4 ov;
    ov.x = xv.x * gv.x; ov.y = xv.y * gv.y;
    ov.z = xv.z * gv.z; ov.w = xv.w * gv.w;
    o4[i] = ov;
}

extern "C" void kernel_launch(void* const* d_in, const int* in_sizes, int n_in,
                              void* d_out, int out_size, void* d_ws, size_t ws_size,
                              hipStream_t stream) {
    (void)in_sizes; (void)n_in; (void)out_size; (void)ws_size;
    const float* x     = (const float*)d_in[0];
    const float* w_sq  = (const float*)d_in[1];
    const float* wq    = (const float*)d_in[2];
    const float* bq    = (const float*)d_in[3];
    const float* wk    = (const float*)d_in[4];
    const float* bk    = (const float*)d_in[5];
    const float* wv    = (const float*)d_in[6];
    const float* bv    = (const float*)d_in[7];
    const float* gamma = (const float*)d_in[8];
    float* out = (float*)d_out;

    float*    z    = (float*)d_ws;                       // B*HW
    float*    gate = z + BATCH * HW;                     // B*HW
    unsigned* zmax = (unsigned*)(gate + BATCH * HW);     // B
    unsigned* zmin = zmax + BATCH;                       // B

    init_kernel<<<1, 32, 0, stream>>>(zmax, zmin);
    squeeze_kernel<<<dim3(HW / 256, BATCH), 256, 0, stream>>>(x, w_sq, z, zmax, zmin);
    attn_kernel<<<dim3(HW / 128, BATCH), 256, 0, stream>>>(
        z, wq, bq, wk, bk, wv, bv, gamma, zmax, zmin, gate);
    const int total4 = BATCH * C_CH * HW / 4;
    apply_kernel<<<total4 / 256, 256, 0, stream>>>(
        (const float4*)x, (const float4*)gate, (float4*)out);
}
